// MultiHeadAttention_9861244912223
// MI455X (gfx1250) — compile-verified, matrix operand order checked by reading, not run on hardware
//
#include <hip/hip_runtime.h>
#include <hip/hip_bf16.h>

typedef __attribute__((ext_vector_type(16))) _Float16 v16h;
typedef __attribute__((ext_vector_type(8)))  float    v8f;
typedef __attribute__((ext_vector_type(4)))  _Float16 h4;

// exact pointee type from the builtin's diagnostic: vector_size(16) int
typedef int v4i_b128 __attribute__((vector_size(4 * sizeof(int))));
typedef __attribute__((address_space(1))) v4i_b128* gptr_b128;  // global (__device__)
typedef __attribute__((address_space(3))) v4i_b128* lptr_b128;  // LDS (__shared__)

union FragU { v16h v; uint4 u[2]; };
union H8U   { uint4 u; _Float16 h[8]; };

#define GM 8192
#define GK 1024
#define GN 1024
#define BK 32
#define NKT (GK / BK)
#define LDT 40      // GEMM LDS row stride (halves): 80B, 16B-aligned, conflict-free
#define SLEN 1024
#define DH 64
#define LDK 72
#define LDV 40
#define LDP 40

#if __has_builtin(__builtin_amdgcn_global_load_async_to_lds_b128)
#define HAS_ASYNC 1
#else
#define HAS_ASYNC 0
#endif

// ---- async global -> LDS copy (ASYNCcnt path), per-lane 16B ----
__device__ __forceinline__ void cp_async_b128(void* lds_dst, const void* gsrc) {
#if HAS_ASYNC
  __builtin_amdgcn_global_load_async_to_lds_b128(
      (gptr_b128)(uintptr_t)gsrc,
      (lptr_b128)(unsigned int)(uintptr_t)lds_dst,
      0, 0);
#else
  *(uint4*)lds_dst = *(const uint4*)gsrc;
#endif
}

__device__ __forceinline__ void async_wait0() {
#if HAS_ASYNC
#if __has_builtin(__builtin_amdgcn_s_wait_asynccnt)
  __builtin_amdgcn_s_wait_asynccnt(0);
#else
  asm volatile("s_wait_asynccnt 0" ::: "memory");
#endif
#endif
}

// ---- XOR butterfly within 16-lane groups: v_permlane16_b32 (pure VALU) ----
__device__ __forceinline__ float lane16_xor(float v, unsigned s1, unsigned s2, int mask) {
#if __has_builtin(__builtin_amdgcn_permlane16)
  unsigned u = __builtin_bit_cast(unsigned, v);
  u = __builtin_amdgcn_permlane16(u, u, s1, s2, false, false);
  (void)mask;
  return __builtin_bit_cast(float, u);
#else
  (void)s1; (void)s2;
  return __shfl_xor(v, mask, 32);
#endif
}
__device__ __forceinline__ float red_max16(float v) {
  v = fmaxf(v, lane16_xor(v, 0x67452301u, 0xEFCDAB89u, 1));
  v = fmaxf(v, lane16_xor(v, 0x54761032u, 0xDCFE98BAu, 2));
  v = fmaxf(v, lane16_xor(v, 0x32107654u, 0xBA98FEDCu, 4));
  v = fmaxf(v, lane16_xor(v, 0xFEDCBA98u, 0x76543210u, 8));
  return v;
}
__device__ __forceinline__ float red_sum16(float v) {
  v += lane16_xor(v, 0x67452301u, 0xEFCDAB89u, 1);
  v += lane16_xor(v, 0x54761032u, 0xDCFE98BAu, 2);
  v += lane16_xor(v, 0x32107654u, 0xBA98FEDCu, 4);
  v += lane16_xor(v, 0xFEDCBA98u, 0x76543210u, 8);
  return v;
}

// Load one 16x32 f16 fragment (A layout; B mirrors it with N as "row").
// Lane L: row = L&15, kb = (L>>4)*8; halves 0..7 = K kb..kb+7, 8..15 = K kb+16..kb+23.
__device__ __forceinline__ v16h frag_ld(const _Float16* base, int stride, int koff) {
  const int lane = threadIdx.x & 31;
  const int row  = lane & 15;
  const int kb   = (lane >> 4) << 3;
  const _Float16* p = base + row * stride + koff + kb;
  FragU f;
  f.u[0] = *(const uint4*)(p);
  f.u[1] = *(const uint4*)(p + 16);
  return f.v;
}

__device__ __forceinline__ v8f wmma16(v16h a, v16h b, v8f c) {
  return __builtin_amdgcn_wmma_f32_16x16x32_f16(false, a, false, b, (short)0, c, false, false);
}

// one-time f32 -> f16 conversion (weights)
__global__ __launch_bounds__(256)
void cvt_f32f16(const float* __restrict__ in, _Float16* __restrict__ out, int n) {
  const int i = (blockIdx.x * 256 + threadIdx.x) * 4;
  if (i + 3 < n) {
    const float4 v = *(const float4*)(in + i);
    h4 hv; hv[0] = (_Float16)v.x; hv[1] = (_Float16)v.y;
           hv[2] = (_Float16)v.z; hv[3] = (_Float16)v.w;
    *(h4*)(out + i) = hv;
  }
}

// C = A[M,K] @ B[N,K]^T + bias.  B is pre-converted f16.  Double-buffered LDS,
// async global->LDS staging where no conversion is needed.
template<bool A_HALF, bool OUT_HEAD>
__global__ __launch_bounds__(256)
void gemm_wmma(const void* __restrict__ A_, const _Float16* __restrict__ Bw,
               const float* __restrict__ bias, void* __restrict__ out, float alpha)
{
  __shared__ __align__(16) _Float16 ldsA[2][128 * LDT];
  __shared__ __align__(16) _Float16 ldsB[2][128 * LDT];

  const int t    = threadIdx.x;
  const int wave = t >> 5;
  const int lane = t & 31;
  const int wm   = wave >> 2;          // 0..1 (64-row strip)
  const int wn   = wave & 3;           // 0..3 (32-col strip)
  const int m0   = blockIdx.y * 128;
  const int n0   = blockIdx.x * 128;

  const v8f vzero = {0.f,0.f,0.f,0.f,0.f,0.f,0.f,0.f};
  v8f acc[4][2];
#pragma unroll
  for (int i = 0; i < 4; ++i)
#pragma unroll
    for (int j = 0; j < 2; ++j) acc[i][j] = vzero;

  auto stage = [&](int kt, int buf) {
    const int k0 = kt * BK;
    if (A_HALF) {
      const _Float16* A = (const _Float16*)A_;
#pragma unroll
      for (int j = 0; j < 2; ++j) {
        const int idx = t + j * 256;               // 512 chunks of 8 halves
        const int r = idx >> 2, c8 = (idx & 3) << 3;
        cp_async_b128(&ldsA[buf][r * LDT + c8],
                      &A[(size_t)(m0 + r) * GK + k0 + c8]);
      }
    } else {
      const float* A = (const float*)A_;
#pragma unroll
      for (int j = 0; j < 4; ++j) {
        const int idx = t + j * 256;               // 1024 chunks of 4 floats
        const int r = idx >> 3, c4 = (idx & 7) << 2;
        const float4 v = *(const float4*)(&A[(size_t)(m0 + r) * GK + k0 + c4]);
        h4 hv; hv[0] = (_Float16)v.x; hv[1] = (_Float16)v.y;
               hv[2] = (_Float16)v.z; hv[3] = (_Float16)v.w;
        *(h4*)(&ldsA[buf][r * LDT + c4]) = hv;
      }
    }
#pragma unroll
    for (int j = 0; j < 2; ++j) {
      const int idx = t + j * 256;
      const int r = idx >> 2, c8 = (idx & 3) << 3;
      cp_async_b128(&ldsB[buf][r * LDT + c8],
                    &Bw[(size_t)(n0 + r) * GK + k0 + c8]);
    }
  };

  stage(0, 0);
  async_wait0();
  __syncthreads();

  for (int kt = 0; kt < NKT; ++kt) {
    const int cur = kt & 1;
    if (kt + 1 < NKT) stage(kt + 1, cur ^ 1);   // overlap next tile with WMMAs

    v16h af[4], bf[2];
#pragma unroll
    for (int mf = 0; mf < 4; ++mf)
      af[mf] = frag_ld(&ldsA[cur][(wm * 64 + mf * 16) * LDT], LDT, 0);
#pragma unroll
    for (int nf = 0; nf < 2; ++nf)
      bf[nf] = frag_ld(&ldsB[cur][(wn * 32 + nf * 16) * LDT], LDT, 0);
#pragma unroll
    for (int mf = 0; mf < 4; ++mf)
#pragma unroll
      for (int nf = 0; nf < 2; ++nf)
        acc[mf][nf] = wmma16(af[mf], bf[nf], acc[mf][nf]);

    async_wait0();
    __syncthreads();
  }

  // epilogue: C layout lane (L&15)=col, VGPR i + 8*(L>=16) = row
  const int colL = lane & 15;
  const int hs   = lane >> 4;
#pragma unroll
  for (int mf = 0; mf < 4; ++mf) {
#pragma unroll
    for (int nf = 0; nf < 2; ++nf) {
      const int n = n0 + wn * 32 + nf * 16 + colL;
      const float bv = bias[n];
#pragma unroll
      for (int i = 0; i < 8; ++i) {
        const int m = m0 + wm * 64 + mf * 16 + i + hs * 8;
        const float val = acc[mf][nf][i] + bv;
        if (OUT_HEAD) {
          _Float16* o = (_Float16*)out;
          const int b = m >> 10, s = m & 1023;   // S = 1024
          const int h = n >> 6,  d = n & 63;     // D = 64
          o[(((size_t)(b * 16 + h) << 10) + s) * 64 + d] = (_Float16)(val * alpha);
        } else {
          float* o = (float*)out;
          o[(size_t)m * GN + n] = val;
        }
      }
    }
  }
}

// Flash-style attention per (batch, head): 128 queries/block, 16/wave,
// 32-key chunks, double-buffered K/V staging (K async).
__global__ __launch_bounds__(256)
void attn_wmma(const _Float16* __restrict__ Qh, const _Float16* __restrict__ Kh,
               const _Float16* __restrict__ Vh, _Float16* __restrict__ Ah)
{
  __shared__ __align__(16) _Float16 ldsK[2][32 * LDK];     // [key][dim]
  __shared__ __align__(16) _Float16 ldsV[2][DH * LDV];     // [dim][key]
  __shared__ __align__(16) _Float16 ldsP[8 * 16 * LDP];    // per-wave P tiles

  const int t    = threadIdx.x;
  const int wave = t >> 5;
  const int lane = t & 31;
  const int colL = lane & 15;
  const int hs   = lane >> 4;

  const int h  = blockIdx.y;
  const int b  = blockIdx.z;
  const int q0 = blockIdx.x * 128 + wave * 16;

  const size_t bh = ((size_t)b * 16 + h) * SLEN * DH;
  const _Float16* Qb = Qh + bh;
  const _Float16* Kb = Kh + bh;
  const _Float16* Vb = Vh + bh;

  // Q fragments straight from global (rows contiguous in D)
  v16h qf[2];
  {
    const int qr = q0 + colL;
    const int kb = hs << 3;
#pragma unroll
    for (int ks = 0; ks < 2; ++ks) {
      FragU f;
      const _Float16* p = Qb + (size_t)qr * DH + ks * 32 + kb;
      f.u[0] = *(const uint4*)(p);
      f.u[1] = *(const uint4*)(p + 16);
      qf[ks] = f.v;
    }
  }

  const v8f vzero = {0.f,0.f,0.f,0.f,0.f,0.f,0.f,0.f};
  v8f o[4];
#pragma unroll
  for (int j = 0; j < 4; ++j) o[j] = vzero;
  float rm[8], rl[8];
#pragma unroll
  for (int i = 0; i < 8; ++i) { rm[i] = -3.0e38f; rl[i] = 0.f; }

  _Float16* pW = &ldsP[wave * 16 * LDP];

  auto stageKV = [&](int kc, int buf) {
    const int kg0 = kc * 32;
    const int key = t >> 3;
    const int c8  = (t & 7) << 3;
    cp_async_b128(&ldsK[buf][key * LDK + c8],
                  &Kb[(size_t)(kg0 + key) * DH + c8]);
    H8U vv; vv.u = *(const uint4*)(&Vb[(size_t)(kg0 + key) * DH + c8]);
#pragma unroll
    for (int j = 0; j < 8; ++j) ldsV[buf][(c8 + j) * LDV + key] = vv.h[j];
  };

  stageKV(0, 0);
  async_wait0();
  __syncthreads();

  for (int kc = 0; kc < SLEN / 32; ++kc) {
    const int cur = kc & 1;
    if (kc + 1 < SLEN / 32) stageKV(kc + 1, cur ^ 1);

    // scores: 16 queries x 32 keys, K = D = 64 via two chained WMMAs per tile
    v8f sc[2];
#pragma unroll
    for (int nf = 0; nf < 2; ++nf) {
      const v16h kf0 = frag_ld(&ldsK[cur][nf * 16 * LDK], LDK, 0);
      const v16h kf1 = frag_ld(&ldsK[cur][nf * 16 * LDK], LDK, 32);
      v8f s = wmma16(qf[0], kf0, vzero);
      sc[nf] = wmma16(qf[1], kf1, s);
    }

    // online softmax; rows live at (vgpr i, lane-half), cols across 16 lanes
    float pv[2][8];
#pragma unroll
    for (int i = 0; i < 8; ++i) {
      const float mx = red_max16(fmaxf(sc[0][i], sc[1][i]));
      const float nm = fmaxf(rm[i], mx);
      const float rescale = __expf(rm[i] - nm);
      const float e0 = __expf(sc[0][i] - nm);
      const float e1 = __expf(sc[1][i] - nm);
      pv[0][i] = e0; pv[1][i] = e1;
      const float rs = red_sum16(e0 + e1);
      rl[i] = rl[i] * rescale + rs;
      rm[i] = nm;
      o[0][i] *= rescale; o[1][i] *= rescale;
      o[2][i] *= rescale; o[3][i] *= rescale;
    }

    // C-layout -> A-layout for P via per-wave LDS tile
#pragma unroll
    for (int nf = 0; nf < 2; ++nf)
#pragma unroll
      for (int i = 0; i < 8; ++i)
        pW[(i + hs * 8) * LDP + nf * 16 + colL] = (_Float16)pv[nf][i];

    asm volatile("s_wait_dscnt 0" ::: "memory");  // intra-wave LDS RAW

    const v16h pf = frag_ld(pW, LDP, 0);
#pragma unroll
    for (int j = 0; j < 4; ++j) {
      const v16h vf = frag_ld(&ldsV[cur][j * 16 * LDV], LDV, 0);
      o[j] = wmma16(pf, vf, o[j]);
    }

    async_wait0();
    __syncthreads();
  }

  // epilogue: O /= l, f16, [n*S + q][E] layout for the final GEMM
#pragma unroll
  for (int j = 0; j < 4; ++j) {
    const int d = j * 16 + colL;
#pragma unroll
    for (int i = 0; i < 8; ++i) {
      const int q = q0 + i + hs * 8;
      Ah[((size_t)(b * SLEN + q)) * 1024 + h * DH + d] = (_Float16)(o[j][i] / rl[i]);
    }
  }
}

extern "C" void kernel_launch(void* const* d_in, const int* in_sizes, int n_in,
                              void* d_out, int out_size, void* d_ws, size_t ws_size,
                              hipStream_t stream) {
  (void)in_sizes; (void)n_in; (void)out_size; (void)ws_size;
  const float* queries = (const float*)d_in[0];
  const float* keys    = (const float*)d_in[1];
  const float* values  = (const float*)d_in[2];
  const float* Wq = (const float*)d_in[3];
  const float* bq = (const float*)d_in[4];
  const float* Wk = (const float*)d_in[5];
  const float* bk = (const float*)d_in[6];
  const float* Wv = (const float*)d_in[7];
  const float* bv = (const float*)d_in[8];
  const float* Wo = (const float*)d_in[9];
  const float* bo = (const float*)d_in[10];

  const size_t PROJ = (size_t)8 * 1024 * 1024;   // f16 elements per projected tensor
  const size_t WEL  = (size_t)1024 * 1024;       // f16 elements per weight matrix
  _Float16* Qh  = (_Float16*)d_ws;
  _Float16* Kh  = Qh + PROJ;
  _Float16* Vh  = Kh + PROJ;
  _Float16* Ah  = Vh + PROJ;
  _Float16* Wqh = Ah + PROJ;
  _Float16* Wkh = Wqh + WEL;
  _Float16* Wvh = Wkh + WEL;
  _Float16* Woh = Wvh + WEL;

  const dim3 bb(256);
  const dim3 cg(WEL / 1024);
  cvt_f32f16<<<cg, bb, 0, stream>>>(Wq, Wqh, (int)WEL);
  cvt_f32f16<<<cg, bb, 0, stream>>>(Wk, Wkh, (int)WEL);
  cvt_f32f16<<<cg, bb, 0, stream>>>(Wv, Wvh, (int)WEL);
  cvt_f32f16<<<cg, bb, 0, stream>>>(Wo, Woh, (int)WEL);

  const dim3 gg(GN / 128, GM / 128);   // 8 x 64

  // projections; 1/sqrt(E) = 1/32 folded into Q
  gemm_wmma<false, true><<<gg, bb, 0, stream>>>(queries, Wqh, bq, Qh, 1.0f / 32.0f);
  gemm_wmma<false, true><<<gg, bb, 0, stream>>>(keys,    Wkh, bk, Kh, 1.0f);
  gemm_wmma<false, true><<<gg, bb, 0, stream>>>(values,  Wvh, bv, Vh, 1.0f);

  attn_wmma<<<dim3(SLEN / 128, 16, 8), bb, 0, stream>>>(Qh, Kh, Vh, Ah);

  // output projection, f32 out + bias
  gemm_wmma<true, false><<<gg, bb, 0, stream>>>(Ah, Woh, bo, (float*)d_out, 1.0f);
}
